// GINMultiTask_33921651703946
// MI455X (gfx1250) — matrix operand order repaired and checked by reading, not compiled, not run on hardware
//
#include <hip/hip_runtime.h>
#include <hip/hip_bf16.h>

typedef __attribute__((ext_vector_type(16))) _Float16 v16h;
typedef __attribute__((ext_vector_type(8)))  _Float16 v8h;
typedef __attribute__((ext_vector_type(8)))  float    v8f;

#define D       300   // embedding dim
#define DP      304   // fp32 row pad (19*16)
#define DK      320   // f16 K pad for GEMM1 (10*32)
#define H2      600   // hidden dim
#define NT1P    40    // GEMM1 col tiles padded to 5 per wave (640 cols)
#define T1      5     // tiles per wave, GEMM1
#define H2K     640   // f16 K pad for GEMM2 (20*32)
#define NT2P    24    // GEMM2 col tiles padded to 3 per wave (384 cols)
#define T2      3     // tiles per wave, GEMM2
#define MROWS   32    // M rows per block (2 row-subtiles of 16)
#define LDSS    648   // LDS row stride in halfs (bank-conflict pad)
#define NTASKS  12
#define BN_EPS  1e-5f

// ---------------- node embedding ----------------
__global__ void node_embed_kernel(const int* __restrict__ x,
                                  const float* __restrict__ e1,
                                  const float* __restrict__ e2,
                                  float* __restrict__ h, int N) {
  int c = threadIdx.x;            // 320 threads
  int r = blockIdx.x;             // N blocks
  if (c >= DP) return;
  int x0 = x[2 * r], x1 = x[2 * r + 1];
  float v = (c < D) ? e1[(size_t)x0 * D + c] + e2[(size_t)x1 * D + c] : 0.f;
  h[(size_t)r * DP + c] = v;
}

// ---------------- per-layer weight prep (transpose + f16, tile-padded) ----------------
__global__ void prep_w1t_kernel(const float* __restrict__ W1, _Float16* __restrict__ w1t) {
  int k = threadIdx.x;            // 320
  int n = blockIdx.x;             // 640 (NT1P*16)
  float v = (k < D && n < H2) ? W1[(size_t)k * H2 + n] : 0.f;
  w1t[(size_t)n * DK + k] = (_Float16)v;
}

__global__ void prep_w2t_kernel(const float* __restrict__ W2, _Float16* __restrict__ w2t) {
  int k = threadIdx.x;            // 640
  int n = blockIdx.x;             // 384 (NT2P*16)
  float v = (k < H2 && n < D) ? W2[(size_t)k * D + n] : 0.f;
  w2t[(size_t)n * H2K + k] = (_Float16)v;
}

__global__ void prep_selfloop_kernel(const float* __restrict__ ew1,
                                     const float* __restrict__ ew2,
                                     float* __restrict__ sl) {
  int c = threadIdx.x;            // 320
  if (c < DP) sl[c] = (c < D) ? ew1[4 * D + c] + ew2[0 * D + c] : 0.f;
}

// ---------------- agg = h + self_loop ----------------
__global__ void agg_init_kernel(const float* __restrict__ h,
                                const float* __restrict__ sl,
                                float* __restrict__ agg, int N) {
  int c = threadIdx.x;            // 320
  int r = blockIdx.x;             // N
  if (c >= DP) return;
  float v = (c < D) ? h[(size_t)r * DP + c] + sl[c] : 0.f;
  agg[(size_t)r * DP + c] = v;
}

// ---------------- scatter-add messages ----------------
__global__ void scatter_kernel(const int* __restrict__ ei, const int* __restrict__ ea,
                               const float* __restrict__ ew1, const float* __restrict__ ew2,
                               const float* __restrict__ h, float* __restrict__ agg, int E) {
  int c = threadIdx.x;            // 320
  int e = blockIdx.x;             // E
  if (c >= D) return;
  int s  = ei[e];
  int d2 = ei[E + e];
  int a0 = ea[2 * e], a1 = ea[2 * e + 1];
  float msg = h[(size_t)s * DP + c] + ew1[(size_t)a0 * D + c] + ew2[(size_t)a1 * D + c];
  atomicAdd(&agg[(size_t)d2 * DP + c], msg);
}

// ---------------- agg(f32) -> A matrix (f16, K-padded) ----------------
__global__ void a16_kernel(const float* __restrict__ agg, _Float16* __restrict__ a16, int N) {
  int k = threadIdx.x;            // 320
  int r = blockIdx.x;             // Npad rows
  float v = (r < N && k < D) ? agg[(size_t)r * DP + k] : 0.f;
  a16[(size_t)r * DK + k] = (_Float16)v;
}

// ---------------- fused MLP: GEMM1 -> ReLU -> GEMM2 (WMMA f16 / f32 acc) ----------------
// Block = 256 threads = 8 waves; M=32 rows per block (2 row-subtiles reuse each
// B fragment). Wave w owns col tiles {w, w+8, ...}: 5 tiles GEMM1, 3 tiles GEMM2.
// B fragments in distinct registers (bf[]) so loads clause-batch ahead of WMMAs.
__global__ __launch_bounds__(256)
void mlp_wmma_kernel(const _Float16* __restrict__ a16,
                     const _Float16* __restrict__ w1t,
                     const _Float16* __restrict__ w2t,
                     const float* __restrict__ b1,
                     const float* __restrict__ b2,
                     float* __restrict__ hpre, int N) {
  __shared__ __align__(16) _Float16 lds_t[MROWS * LDSS];
  const int tid  = threadIdx.x;
  const int w    = tid >> 5;      // wave id 0..7
  const int lane = tid & 31;
  const int m    = lane & 15;     // row (A) / col-in-tile (B,C)
  const int hi   = lane >> 4;     // half-wave select
  const int row0 = blockIdx.x * MROWS;

  // ---- GEMM1: [32 x 320] x [320 x 640(pad)] ----
  v8f acc1[T1][2] = {};
  const _Float16* arow0 = a16 + (size_t)(row0 + m) * DK + hi * 16;
  const _Float16* arow1 = arow0 + (size_t)16 * DK;
  const _Float16* b1base = w1t + (size_t)(w * 16 + m) * DK + hi * 16;
  for (int kc = 0; kc < DK; kc += 32) {
    v8h a0lo = *(const v8h*)(arow0 + kc);
    v8h a0hi = *(const v8h*)(arow0 + kc + 8);
    v8h a1lo = *(const v8h*)(arow1 + kc);
    v8h a1hi = *(const v8h*)(arow1 + kc + 8);
    v16h af0 = __builtin_shufflevector(a0lo, a0hi, 0,1,2,3,4,5,6,7,8,9,10,11,12,13,14,15);
    v16h af1 = __builtin_shufflevector(a1lo, a1hi, 0,1,2,3,4,5,6,7,8,9,10,11,12,13,14,15);
    v16h bf[T1];
#pragma unroll
    for (int t = 0; t < T1; ++t)
      bf[t] = *(const v16h*)(b1base + (size_t)(t * 8 * 16) * DK + kc);
#pragma unroll
    for (int t = 0; t < T1; ++t) {
      acc1[t][0] = __builtin_amdgcn_wmma_f32_16x16x32_f16(
          false, af0, false, bf[t], (short)0, acc1[t][0], false, false);
      acc1[t][1] = __builtin_amdgcn_wmma_f32_16x16x32_f16(
          false, af1, false, bf[t], (short)0, acc1[t][1], false, false);
    }
  }

  // bias + ReLU + store t-tiles into LDS (A layout source for GEMM2).
  // Padded tiles (cols 608..639) store exact zeros == required GEMM2 K-pad.
#pragma unroll
  for (int t = 0; t < T1; ++t) {
    int col = (w + 8 * t) * 16 + m;
    float bias = (col < H2) ? b1[col] : 0.f;
#pragma unroll
    for (int s = 0; s < 2; ++s) {
#pragma unroll
      for (int r = 0; r < 8; ++r) {
        float v = acc1[t][s][r] + bias;
        v = v > 0.f ? v : 0.f;
        lds_t[(s * 16 + r + 8 * hi) * LDSS + col] = (_Float16)v;
      }
    }
  }
  __syncthreads();

  // ---- GEMM2: [32 x 640] x [640 x 384(pad)] ----
  v8f acc2[T2][2] = {};
  const _Float16* lrow0 = lds_t + m * LDSS + hi * 16;
  const _Float16* lrow1 = lrow0 + (size_t)16 * LDSS;
  const _Float16* b2base = w2t + (size_t)(w * 16 + m) * H2K + hi * 16;
  for (int kc = 0; kc < H2K; kc += 32) {
    v8h a0lo = *(const v8h*)(lrow0 + kc);
    v8h a0hi = *(const v8h*)(lrow0 + kc + 8);
    v8h a1lo = *(const v8h*)(lrow1 + kc);
    v8h a1hi = *(const v8h*)(lrow1 + kc + 8);
    v16h af0 = __builtin_shufflevector(a0lo, a0hi, 0,1,2,3,4,5,6,7,8,9,10,11,12,13,14,15);
    v16h af1 = __builtin_shufflevector(a1lo, a1hi, 0,1,2,3,4,5,6,7,8,9,10,11,12,13,14,15);
    v16h bf[T2];
#pragma unroll
    for (int t = 0; t < T2; ++t)
      bf[t] = *(const v16h*)(b2base + (size_t)(t * 8 * 16) * H2K + kc);
#pragma unroll
    for (int t = 0; t < T2; ++t) {
      acc2[t][0] = __builtin_amdgcn_wmma_f32_16x16x32_f16(
          false, af0, false, bf[t], (short)0, acc2[t][0], false, false);
      acc2[t][1] = __builtin_amdgcn_wmma_f32_16x16x32_f16(
          false, af1, false, bf[t], (short)0, acc2[t][1], false, false);
    }
  }

  // bias + store pre-BN result (guards only in epilogue)
#pragma unroll
  for (int t = 0; t < T2; ++t) {
    int col = (w + 8 * t) * 16 + m;
    if (col < D) {
      float bias = b2[col];
#pragma unroll
      for (int s = 0; s < 2; ++s) {
#pragma unroll
        for (int r = 0; r < 8; ++r) {
          int row = row0 + s * 16 + r + 8 * hi;
          if (row < N) hpre[(size_t)row * DP + col] = acc2[t][s][r] + bias;
        }
      }
    }
  }
}

// ---------------- BatchNorm (training-mode batch stats) ----------------
__global__ void zero_kernel(float* __restrict__ p, size_t n) {
  size_t i = (size_t)blockIdx.x * blockDim.x + threadIdx.x;
  if (i < n) p[i] = 0.f;
}

__global__ void bn_stats_kernel(const float* __restrict__ hpre,
                                float* __restrict__ sum, float* __restrict__ sumsq, int N) {
  int c = threadIdx.x;            // 320
  if (c >= D) return;
  float s = 0.f, q = 0.f;
  for (int r = blockIdx.x; r < N; r += gridDim.x) {
    float v = hpre[(size_t)r * DP + c];
    s += v; q += v * v;
  }
  atomicAdd(&sum[c], s);
  atomicAdd(&sumsq[c], q);
}

__global__ void bn_apply_kernel(const float* __restrict__ hpre,
                                const float* __restrict__ sum, const float* __restrict__ sumsq,
                                const float* __restrict__ g, const float* __restrict__ b,
                                float* __restrict__ h, int N, int do_relu) {
  int c = threadIdx.x;            // 320
  if (c >= D) return;
  float mu  = sum[c] / (float)N;
  float var = sumsq[c] / (float)N - mu * mu;
  float sc  = rsqrtf(var + BN_EPS) * g[c];
  float bb  = b[c] - mu * sc;
  for (int r = blockIdx.x; r < N; r += gridDim.x) {
    float v = hpre[(size_t)r * DP + c] * sc + bb;
    if (do_relu) v = v > 0.f ? v : 0.f;
    h[(size_t)r * DP + c] = v;
  }
}

// ---------------- global mean pool + head ----------------
__global__ void pool_kernel(const float* __restrict__ h, const int* __restrict__ batch,
                            float* __restrict__ psum, float* __restrict__ pcnt, int N) {
  int c = threadIdx.x;            // 320
  int r = blockIdx.x;             // N
  int g = batch[r];
  if (c < D) atomicAdd(&psum[(size_t)g * DP + c], h[(size_t)r * DP + c]);
  if (c == 0) atomicAdd(&pcnt[g], 1.f);
}

__global__ void head_kernel(const float* __restrict__ psum, const float* __restrict__ pcnt,
                            const float* __restrict__ hW, const float* __restrict__ hb,
                            float* __restrict__ out, int G) {
  int idx = blockIdx.x * blockDim.x + threadIdx.x;
  if (idx >= G * NTASKS) return;
  int g = idx / NTASKS, t = idx % NTASKS;
  float inv = 1.f / fmaxf(pcnt[g], 1.f);
  float acc = hb[t];
  for (int d = 0; d < D; ++d)
    acc += psum[(size_t)g * DP + d] * inv * hW[d * NTASKS + t];
  out[idx] = acc;
}

// ---------------- driver ----------------
extern "C" void kernel_launch(void* const* d_in, const int* in_sizes, int n_in,
                              void* d_out, int out_size, void* d_ws, size_t ws_size,
                              hipStream_t stream) {
  const int*   x      = (const int*)d_in[0];
  const int*   ei     = (const int*)d_in[1];
  const int*   ea     = (const int*)d_in[2];
  const int*   batch  = (const int*)d_in[3];
  const float* atom1  = (const float*)d_in[4];
  const float* atom2  = (const float*)d_in[5];
  const float* edgew1 = (const float*)d_in[6];
  const float* edgew2 = (const float*)d_in[7];
  const float* W1     = (const float*)d_in[8];
  const float* b1     = (const float*)d_in[9];
  const float* W2     = (const float*)d_in[10];
  const float* b2     = (const float*)d_in[11];
  const float* bn_g   = (const float*)d_in[12];
  const float* bn_b   = (const float*)d_in[13];
  const float* headW  = (const float*)d_in[14];
  const float* headb  = (const float*)d_in[15];

  const int N      = in_sizes[3];
  const int E      = in_sizes[1] / 2;
  const int G      = out_size / NTASKS;
  const int Mtiles = (N + MROWS - 1) / MROWS;
  const int Npad   = Mtiles * MROWS;

  char* ws = (char*)d_ws;
  size_t off = 0;
  auto alloc = [&](size_t bytes) -> char* {
    char* p = ws + off;
    off = (off + bytes + 255) & ~(size_t)255;
    return p;
  };
  float*    h    = (float*)alloc((size_t)N * DP * sizeof(float));
  float*    agg  = (float*)alloc((size_t)N * DP * sizeof(float));  // also reused as hpre
  _Float16* a16  = (_Float16*)alloc((size_t)Npad * DK * sizeof(_Float16));
  _Float16* w1t  = (_Float16*)alloc((size_t)(NT1P * 16) * DK * sizeof(_Float16));
  _Float16* w2t  = (_Float16*)alloc((size_t)(NT2P * 16) * H2K * sizeof(_Float16));
  float*    sl   = (float*)alloc(DP * sizeof(float));
  float*    bsum = (float*)alloc(DP * sizeof(float));
  float*    bsq  = (float*)alloc(DP * sizeof(float));
  float*    psum = (float*)alloc((size_t)G * DP * sizeof(float));
  float*    pcnt = (float*)alloc((size_t)G * sizeof(float));
  (void)ws_size; (void)n_in;

  node_embed_kernel<<<N, 320, 0, stream>>>(x, atom1, atom2, h, N);

  for (int l = 0; l < 5; ++l) {
    const float* ew1 = edgew1 + (size_t)l * 6 * D;
    const float* ew2 = edgew2 + (size_t)l * 3 * D;
    prep_w1t_kernel<<<NT1P * 16, DK, 0, stream>>>(W1 + (size_t)l * D * H2, w1t);
    prep_w2t_kernel<<<NT2P * 16, H2K, 0, stream>>>(W2 + (size_t)l * H2 * D, w2t);
    prep_selfloop_kernel<<<1, 320, 0, stream>>>(ew1, ew2, sl);
    agg_init_kernel<<<N, 320, 0, stream>>>(h, sl, agg, N);
    scatter_kernel<<<E, 320, 0, stream>>>(ei, ea, ew1, ew2, h, agg, E);
    a16_kernel<<<Npad, DK, 0, stream>>>(agg, a16, N);
    mlp_wmma_kernel<<<Mtiles, 256, 0, stream>>>(a16, w1t, w2t,
                                                b1 + (size_t)l * H2,
                                                b2 + (size_t)l * D, agg, N);
    zero_kernel<<<1, 320, 0, stream>>>(bsum, DP);
    zero_kernel<<<1, 320, 0, stream>>>(bsq, DP);
    bn_stats_kernel<<<1024, 320, 0, stream>>>(agg, bsum, bsq, N);
    bn_apply_kernel<<<2048, 320, 0, stream>>>(agg, bsum, bsq,
                                              bn_g + (size_t)l * D, bn_b + (size_t)l * D,
                                              h, N, (l != 4) ? 1 : 0);
  }

  zero_kernel<<<(int)(((size_t)G * DP + 255) / 256), 256, 0, stream>>>(psum, (size_t)G * DP);
  zero_kernel<<<(G + 255) / 256, 256, 0, stream>>>(pcnt, (size_t)G);
  pool_kernel<<<N, 320, 0, stream>>>(h, batch, psum, pcnt, N);
  head_kernel<<<(G * NTASKS + 255) / 256, 256, 0, stream>>>(psum, pcnt, headW, headb,
                                                            (float*)d_out, G);
}
